// SelfAttention_35038343201280
// MI455X (gfx1250) — matrix operand order repaired and checked by reading, not compiled, not run on hardware
//
#include <hip/hip_runtime.h>
#include <hip/hip_bf16.h>
#include <stdint.h>

// Problem constants (reference: B=8, S=2048, D=1024)
#define BATCH 8
#define SEQ   2048
#define DMODEL 1024

typedef unsigned short ushort_t;
typedef __attribute__((ext_vector_type(16))) __bf16 v16bf;
typedef __attribute__((ext_vector_type(8)))  float  v8f;

union FragAB { v16bf v; uint32_t u[8]; };

__device__ __forceinline__ ushort_t f2bf(float f) {
  uint32_t u = __float_as_uint(f);
  u += 0x7FFFu + ((u >> 16) & 1u);          // round-to-nearest-even
  return (ushort_t)(u >> 16);
}

// CDNA5 async global->LDS copy, 16 bytes per lane (per-lane addresses).
__device__ __forceinline__ void async_cp16(uint32_t lds_off, const void* g) {
  asm volatile("global_load_async_to_lds_b128 %0, %1, off"
               :: "v"(lds_off), "v"((unsigned long long)(uintptr_t)g)
               : "memory");
}

__device__ __forceinline__ void wait_async0() {
#if __has_builtin(__builtin_amdgcn_s_wait_asynccnt)
  __builtin_amdgcn_s_wait_asynccnt(0);
#else
  asm volatile("s_wait_asynccnt 0" ::: "memory");
#endif
}

// ---------------------------------------------------------------------------
// LayerNorm over D=1024, one row per block (256 threads, 4 floats/thread).
// ---------------------------------------------------------------------------
__global__ __launch_bounds__(256)
void ln_bf16_kernel(const float* __restrict__ x,
                    const float* __restrict__ gamma,
                    const float* __restrict__ beta,
                    ushort_t* __restrict__ xn) {
  const int row = blockIdx.x;
  const int tid = threadIdx.x;
  const size_t base = (size_t)row * DMODEL;
  const float4 v = ((const float4*)(x + base))[tid];
  float s  = v.x + v.y + v.z + v.w;
  float sq = v.x * v.x + v.y * v.y + v.z * v.z + v.w * v.w;
  for (int o = 16; o > 0; o >>= 1) {
    s  += __shfl_xor(s,  o, 32);
    sq += __shfl_xor(sq, o, 32);
  }
  __shared__ float ps[8], pq[8];
  const int wid = tid >> 5, lane = tid & 31;
  if (lane == 0) { ps[wid] = s; pq[wid] = sq; }
  __syncthreads();
  float ts = 0.f, tq = 0.f;
  for (int i = 0; i < 8; ++i) { ts += ps[i]; tq += pq[i]; }
  const float mu  = ts * (1.0f / DMODEL);
  const float var = tq * (1.0f / DMODEL) - mu * mu;
  const float inv = rsqrtf(var + 1e-5f);
  const int c = tid * 4;
  const float o0 = (v.x - mu) * inv * gamma[c + 0] + beta[c + 0];
  const float o1 = (v.y - mu) * inv * gamma[c + 1] + beta[c + 1];
  const float o2 = (v.z - mu) * inv * gamma[c + 2] + beta[c + 2];
  const float o3 = (v.w - mu) * inv * gamma[c + 3] + beta[c + 3];
  uint2 pk;
  pk.x = (uint32_t)f2bf(o0) | ((uint32_t)f2bf(o1) << 16);
  pk.y = (uint32_t)f2bf(o2) | ((uint32_t)f2bf(o3) << 16);
  ((uint2*)(xn + base))[tid] = pk;
}

// ---------------------------------------------------------------------------
// fp32 -> bf16 elementwise conversion (vectorized x4)
// ---------------------------------------------------------------------------
__global__ __launch_bounds__(256)
void f32_to_bf16_kernel(const float* __restrict__ src,
                        ushort_t* __restrict__ dst, int n4) {
  const int i = blockIdx.x * 256 + threadIdx.x;
  if (i >= n4) return;
  const float4 v = ((const float4*)src)[i];
  uint2 pk;
  pk.x = (uint32_t)f2bf(v.x) | ((uint32_t)f2bf(v.y) << 16);
  pk.y = (uint32_t)f2bf(v.z) | ((uint32_t)f2bf(v.w) << 16);
  ((uint2*)dst)[i] = pk;
}

// ---------------------------------------------------------------------------
// bf16 WMMA GEMM:  C[m,n] = scale * sum_k A[m,k]*B[k,n] (+bias[n]) (+resid)
// B element (k,n) stored at Bm[n*ldb + k]  (i.e. B supplied as [N,K]).
// Block tile 128x128, K-step 32, double-buffered async-to-LDS staging.
// 8 waves (wave32); each wave owns 32x64 = 2x4 v_wmma_f32_16x16x32_bf16 tiles.
// Outputs (each optional): fp32 C, bf16 C, bf16 C-transposed (per-batch).
// All M,N multiples of 128 and K multiples of 32 (guaranteed by caller).
// ---------------------------------------------------------------------------
__global__ __launch_bounds__(256)
void gemm_bf16_wmma(const ushort_t* __restrict__ A, int lda, long long strideA,
                    const ushort_t* __restrict__ Bm, int ldb, long long strideB,
                    const float* __restrict__ bias,
                    float scale,
                    const float* __restrict__ resid, long long strideR,
                    float* __restrict__ outF, int ldc, long long strideF,
                    ushort_t* __restrict__ outH, long long strideH,
                    ushort_t* __restrict__ outHT, int ldcT, long long strideHT,
                    int K) {
  constexpr int BM = 128, BN = 128, BK = 32;
  __shared__ __align__(16) ushort_t Alds[2][BM * BK];
  __shared__ __align__(16) ushort_t Blds[2][BK * BN];

  const int tid  = threadIdx.x;
  const int lane = tid & 31;
  const int wid  = tid >> 5;
  const int half = lane >> 4;     // 0: lanes 0-15, 1: lanes 16-31
  const int l15  = lane & 15;
  const int wm   = wid & 3;       // wave's 32-row slice within 128
  const int wn   = wid >> 2;      // wave's 64-col slice within 128

  const int m0 = blockIdx.y * BM;
  const int n0 = blockIdx.x * BN;
  const int b  = blockIdx.z;
  const ushort_t* Ab = A  + (size_t)b * strideA;
  const ushort_t* Bb = Bm + (size_t)b * strideB;

  v8f acc[2][4];
  for (int i = 0; i < 2; ++i)
    for (int j = 0; j < 4; ++j)
      for (int r = 0; r < 8; ++r) acc[i][j][r] = 0.0f;

  // Straight-line staging: each thread moves 2x16B of A and 2x16B of B per
  // K-step; global pointers advance by BK elements each step.
  const int rT  = tid >> 2;                 // 0..63
  const int kcT = (tid & 3) * 8;            // 0,8,16,24
  const ushort_t* pA0 = Ab + (size_t)(m0 + rT) * lda + kcT;
  const ushort_t* pA1 = pA0 + (size_t)64 * lda;
  const ushort_t* pB0 = Bb + (size_t)(n0 + rT) * ldb + kcT;
  const ushort_t* pB1 = pB0 + (size_t)64 * ldb;
  const uint32_t aOff0 = (uint32_t)(rT * BK + kcT) * 2;
  const uint32_t aOff1 = aOff0 + 64 * BK * 2;
  const uint32_t aBase[2] = {(uint32_t)(uintptr_t)(&Alds[0][0]),
                             (uint32_t)(uintptr_t)(&Alds[1][0])};
  const uint32_t bBase[2] = {(uint32_t)(uintptr_t)(&Blds[0][0]),
                             (uint32_t)(uintptr_t)(&Blds[1][0])};

  auto stage = [&](int buf) {
    async_cp16(aBase[buf] + aOff0, pA0);
    async_cp16(aBase[buf] + aOff1, pA1);
    async_cp16(bBase[buf] + aOff0, pB0);
    async_cp16(bBase[buf] + aOff1, pB1);
    pA0 += BK; pA1 += BK; pB0 += BK; pB1 += BK;
  };

  // Prologue: fill buffer 0.
  stage(0);
  wait_async0();
  __syncthreads();

  const int nSteps = K / BK;
  for (int st = 0; st < nSteps; ++st) {
    const int buf = st & 1;
    if (st + 1 < nSteps) stage(buf ^ 1);   // prefetch next tile (async engine)

    const ushort_t* aP = &Alds[buf][0];
    const ushort_t* bP = &Blds[buf][0];

    // A fragments (ISA 7.12.2, 16-bit A 16x32): k runs 8*half+0..7, 16+8*half+0..7 -> ds_load_b128 x2
    FragAB afr[2];
#pragma unroll
    for (int tm = 0; tm < 2; ++tm) {
      const int row = wm * 32 + tm * 16 + l15;
#pragma unroll
      for (int v = 0; v < 8; ++v) {
        const int kk = (v & 3) * 2 + half * 8 + (v >> 2) * 16;
        afr[tm].u[v] = *(const uint32_t*)&aP[row * BK + kk];
      }
    }
    // B fragments (32x16): lane l15 = column, same K order as A fragments.
    FragAB bfr[4];
#pragma unroll
    for (int tn = 0; tn < 4; ++tn) {
      const int col = wn * 64 + tn * 16 + l15;
#pragma unroll
      for (int v = 0; v < 8; ++v) {
        const int kk = (v & 3) * 2 + half * 8 + (v >> 2) * 16;
        bfr[tn].u[v] = *(const uint32_t*)&bP[col * BK + kk];
      }
    }
#pragma unroll
    for (int tm = 0; tm < 2; ++tm)
#pragma unroll
      for (int tn = 0; tn < 4; ++tn)
        acc[tm][tn] = __builtin_amdgcn_wmma_f32_16x16x32_bf16(
            false, afr[tm].v, false, bfr[tn].v,
            (short)0, acc[tm][tn], false, false);

    wait_async0();     // own prefetches landed
    __syncthreads();   // everyone's prefetches landed; buf readers done
  }

  // Epilogue. C layout (ISA 7.12.2): VGPR r holds row (r + 8*half), col l15.
  const float*   rb  = resid ? resid + (size_t)b * strideR  : nullptr;
  float*         fb  = outF  ? outF  + (size_t)b * strideF  : nullptr;
  ushort_t*      hb  = outH  ? outH  + (size_t)b * strideH  : nullptr;
  ushort_t*      hbT = outHT ? outHT + (size_t)b * strideHT : nullptr;
  for (int tm = 0; tm < 2; ++tm) {
    for (int tn = 0; tn < 4; ++tn) {
      const int col = n0 + wn * 64 + tn * 16 + l15;
      const int rowBase = m0 + wm * 32 + tm * 16 + half * 8;
      const float bv = bias ? bias[col] : 0.0f;
      float vals[8];
#pragma unroll
      for (int r = 0; r < 8; ++r) {
        const size_t idx = (size_t)(rowBase + r) * ldc + col;
        float val = acc[tm][tn][r] * scale + bv;
        if (rb) val += rb[idx];
        vals[r] = val;
        if (fb) fb[idx] = val;
        if (hb) hb[idx] = f2bf(val);
      }
      if (hbT) {
        // Transposed bf16 store: per lane 8 consecutive rows at fixed col
        // -> one 16B store (rowBase is a multiple of 8).
        uint4 pk;
        pk.x = (uint32_t)f2bf(vals[0]) | ((uint32_t)f2bf(vals[1]) << 16);
        pk.y = (uint32_t)f2bf(vals[2]) | ((uint32_t)f2bf(vals[3]) << 16);
        pk.z = (uint32_t)f2bf(vals[4]) | ((uint32_t)f2bf(vals[5]) << 16);
        pk.w = (uint32_t)f2bf(vals[6]) | ((uint32_t)f2bf(vals[7]) << 16);
        *(uint4*)&hbT[(size_t)col * ldcT + rowBase] = pk;
      }
    }
  }
}

// ---------------------------------------------------------------------------
// Row softmax over S=2048: one row per block, 8 elems/thread, all in regs.
// Overwrites fp32 attn in place and writes a bf16 copy for the attn@V GEMM.
// ---------------------------------------------------------------------------
__global__ __launch_bounds__(256)
void softmax_kernel(float* __restrict__ attn, ushort_t* __restrict__ attn_bf) {
  const int row = blockIdx.x;
  const int tid = threadIdx.x;
  float* p = attn + (size_t)row * SEQ;
  const float4 a0 = ((const float4*)p)[tid * 2 + 0];
  const float4 a1 = ((const float4*)p)[tid * 2 + 1];
  float v[8] = {a0.x, a0.y, a0.z, a0.w, a1.x, a1.y, a1.z, a1.w};

  float mx = v[0];
  for (int j = 1; j < 8; ++j) mx = fmaxf(mx, v[j]);
  for (int o = 16; o > 0; o >>= 1) mx = fmaxf(mx, __shfl_xor(mx, o, 32));
  __shared__ float sm[8], ss[8];
  const int wid = tid >> 5, lane = tid & 31;
  if (lane == 0) sm[wid] = mx;
  __syncthreads();
  float gmx = sm[0];
  for (int i = 1; i < 8; ++i) gmx = fmaxf(gmx, sm[i]);

  float s = 0.f;
  for (int j = 0; j < 8; ++j) { v[j] = __expf(v[j] - gmx); s += v[j]; }
  for (int o = 16; o > 0; o >>= 1) s += __shfl_xor(s, o, 32);
  if (lane == 0) ss[wid] = s;
  __syncthreads();
  float tot = 0.f;
  for (int i = 0; i < 8; ++i) tot += ss[i];
  const float rinv = 1.0f / tot;

  float o_[8];
  for (int j = 0; j < 8; ++j) o_[j] = v[j] * rinv;
  float4 w0 = {o_[0], o_[1], o_[2], o_[3]};
  float4 w1 = {o_[4], o_[5], o_[6], o_[7]};
  ((float4*)p)[tid * 2 + 0] = w0;
  ((float4*)p)[tid * 2 + 1] = w1;
  uint4 pk;
  pk.x = (uint32_t)f2bf(o_[0]) | ((uint32_t)f2bf(o_[1]) << 16);
  pk.y = (uint32_t)f2bf(o_[2]) | ((uint32_t)f2bf(o_[3]) << 16);
  pk.z = (uint32_t)f2bf(o_[4]) | ((uint32_t)f2bf(o_[5]) << 16);
  pk.w = (uint32_t)f2bf(o_[6]) | ((uint32_t)f2bf(o_[7]) << 16);
  ((uint4*)(attn_bf + (size_t)row * SEQ))[tid] = pk;
}

// ---------------------------------------------------------------------------
// Host launcher. Workspace layout (bytes), ~198 MB total:
//   Qbf 32M | Kbf 32M | VbfT 32M | xn 32M | attn_bf 64M | Wq/Wk/Wv bf 3x2M
// ---------------------------------------------------------------------------
extern "C" void kernel_launch(void* const* d_in, const int* in_sizes, int n_in,
                              void* d_out, int out_size, void* d_ws, size_t ws_size,
                              hipStream_t stream) {
  (void)in_sizes; (void)n_in; (void)out_size; (void)ws_size;
  const float* x     = (const float*)d_in[0];
  const float* Wq    = (const float*)d_in[1];
  const float* bq    = (const float*)d_in[2];
  const float* Wk    = (const float*)d_in[3];
  const float* bk    = (const float*)d_in[4];
  const float* Wv    = (const float*)d_in[5];
  const float* bv    = (const float*)d_in[6];
  const float* gamma = (const float*)d_in[7];
  const float* beta  = (const float*)d_in[8];

  const size_t SD = (size_t)BATCH * SEQ * DMODEL;   // 16,777,216
  const size_t SS = (size_t)BATCH * SEQ * SEQ;      // 33,554,432
  float* out  = (float*)d_out;       // [B,S,D]
  float* attn = out + SD;            // [B,S,S]

  char* ws = (char*)d_ws;
  ushort_t* Qbf    = (ushort_t*)ws;                 ws += SD * 2;
  ushort_t* Kbf    = (ushort_t*)ws;                 ws += SD * 2;
  ushort_t* VbfT   = (ushort_t*)ws;                 ws += SD * 2;  // [B,D,S]
  ushort_t* xn     = (ushort_t*)ws;                 ws += SD * 2;
  ushort_t* attnbf = (ushort_t*)ws;                 ws += SS * 2;
  ushort_t* Wqbf   = (ushort_t*)ws;                 ws += (size_t)DMODEL * DMODEL * 2;
  ushort_t* Wkbf   = (ushort_t*)ws;                 ws += (size_t)DMODEL * DMODEL * 2;
  ushort_t* Wvbf   = (ushort_t*)ws;

  const long long SDb = (long long)SEQ * DMODEL;  // per-batch elems
  const long long SSb = (long long)SEQ * SEQ;
  const long long DSb = (long long)DMODEL * SEQ;

  // 1) LayerNorm -> bf16 xn
  ln_bf16_kernel<<<BATCH * SEQ, 256, 0, stream>>>(x, gamma, beta, xn);

  // 2) Weights fp32 -> bf16
  const int n4 = (DMODEL * DMODEL) / 4;
  f32_to_bf16_kernel<<<(n4 + 255) / 256, 256, 0, stream>>>(Wq, Wqbf, n4);
  f32_to_bf16_kernel<<<(n4 + 255) / 256, 256, 0, stream>>>(Wk, Wkbf, n4);
  f32_to_bf16_kernel<<<(n4 + 255) / 256, 256, 0, stream>>>(Wv, Wvbf, n4);

  // 3) QKV projections per batch: [2048 x 1024] = xn @ W^T + b (bf16 out).
  //    V is additionally stored transposed ([B,D,S]) so the attn@V GEMM can
  //    use the fast [N,K] B-path.
  {
    dim3 g(DMODEL / 128, SEQ / 128, BATCH);
    gemm_bf16_wmma<<<g, 256, 0, stream>>>(
        xn, DMODEL, SDb, Wqbf, DMODEL, 0, bq, 1.0f,
        nullptr, 0, nullptr, DMODEL, 0,
        Qbf, SDb, nullptr, 0, 0, DMODEL);
    gemm_bf16_wmma<<<g, 256, 0, stream>>>(
        xn, DMODEL, SDb, Wkbf, DMODEL, 0, bk, 1.0f,
        nullptr, 0, nullptr, DMODEL, 0,
        Kbf, SDb, nullptr, 0, 0, DMODEL);
    gemm_bf16_wmma<<<g, 256, 0, stream>>>(
        xn, DMODEL, SDb, Wvbf, DMODEL, 0, bv, 1.0f,
        nullptr, 0, nullptr, DMODEL, 0,
        nullptr, 0, VbfT, SEQ, DSb, DMODEL);
  }

  // 4) scores = (Q @ K^T) / sqrt(D)  -> fp32 into attn region
  {
    dim3 g(SEQ / 128, SEQ / 128, BATCH);
    gemm_bf16_wmma<<<g, 256, 0, stream>>>(
        Qbf, DMODEL, SDb, Kbf, DMODEL, SDb,
        nullptr, 0.03125f /* 1/sqrt(1024) */,
        nullptr, 0,
        attn, SEQ, SSb,
        nullptr, 0, nullptr, 0, 0, DMODEL);
  }

  // 5) softmax rows (fp32 in place + bf16 copy)
  softmax_kernel<<<BATCH * SEQ, 256, 0, stream>>>(attn, attnbf);

  // 6) out = x + attn @ V   (B supplied as V^T, [N=D, K=S])
  {
    dim3 g(DMODEL / 128, SEQ / 128, BATCH);
    gemm_bf16_wmma<<<g, 256, 0, stream>>>(
        attnbf, SEQ, SSb, VbfT, SEQ, DSb,
        nullptr, 1.0f,
        x, SDb,
        out, DMODEL, SDb,
        nullptr, 0, nullptr, 0, 0, SEQ);
  }
}
